// CrossAttention_lucas_52982716564007
// MI455X (gfx1250) — matrix operand order repaired and checked by reading, compiled, not run on hardware
//
#include <hip/hip_runtime.h>

typedef __attribute__((ext_vector_type(16))) _Float16     v16h;
typedef __attribute__((ext_vector_type(8)))  float        v8f;
typedef __attribute__((ext_vector_type(8)))  unsigned int v8u;

#define BATCH 2
#define CHAN  128
#define NSP   4096
#define NHEAD 4
#define DHEAD 32
#define HID   128
#define LOG2E 1.44269504088896340736f

#if defined(__gfx1250__) && __has_builtin(__builtin_amdgcn_tensor_load_to_lds) && \
    __has_builtin(__builtin_amdgcn_s_wait_tensorcnt)
#define USE_TDM 1
#else
#define USE_TDM 0
#endif

__device__ __forceinline__ v16h as_v16h(v8u u) {
  union { v8u u; v16h h; } c; c.u = u; return c.h;
}
__device__ __forceinline__ v16h mk16(uint4 a, uint4 b) {
  union { unsigned int u[8]; v16h h; } c;
  c.u[0] = a.x; c.u[1] = a.y; c.u[2] = a.z; c.u[3] = a.w;
  c.u[4] = b.x; c.u[5] = b.y; c.u[6] = b.z; c.u[7] = b.w;
  return c.h;
}
__device__ __forceinline__ unsigned int pack2h(float a, float b) {
  union { _Float16 h[2]; unsigned int u; } c;
  c.h[0] = (_Float16)a; c.h[1] = (_Float16)b; return c.u;
}

#if USE_TDM
typedef __attribute__((ext_vector_type(4))) unsigned int u32x4;
typedef __attribute__((ext_vector_type(8))) int          i32x8;
typedef __attribute__((ext_vector_type(4))) int          i32x4;

// Issue one TDM 2D tile load: rows x rowlen f16 elements, row stride in
// elements, global tile start -> LDS byte offset. D# per CDNA5 ISA 8.3/8.4.
__device__ __forceinline__ void tdm_load_2d(const _Float16* gsrc,
                                            unsigned lds_off, int rows,
                                            int rowlen, int rowstride) {
  const unsigned long long ga = (unsigned long long)(const void*)gsrc;
  u32x4 g0;
  g0[0] = 1u;                                            // count=1 (valid)
  g0[1] = lds_off;                                       // lds_addr
  g0[2] = (unsigned)(ga & 0xFFFFFFFFull);                // global_addr[31:0]
  g0[3] = (unsigned)((ga >> 32) & 0x1FFFFFFull) | (2u << 30);  // [56:32]|type=2
  i32x8 g1;
  g1[0] = (int)(1u << 16);                               // data_size=2B
  g1[1] = (int)((unsigned)rowlen << 16);                 // tensor_dim0[15:0]
  g1[2] = (int)(((unsigned)rows & 0xFFFFu) << 16);       // tensor_dim1[15:0]
  g1[3] = (int)(((unsigned)rowlen & 0xFFFFu) << 16);     // tile_dim0
  g1[4] = (int)((unsigned)rows & 0xFFFFu);               // tile_dim1
  g1[5] = rowstride;                                     // tensor_dim0_stride
  g1[6] = 0;
  g1[7] = 0;
  const i32x4 z4 = {0, 0, 0, 0};
#if __clang_major__ >= 23
  const i32x8 z8 = {0, 0, 0, 0, 0, 0, 0, 0};
  __builtin_amdgcn_tensor_load_to_lds(g0, g1, z4, z4, z8, 0);
#else
  __builtin_amdgcn_tensor_load_to_lds(g0, g1, z4, z4, 0);
#endif
}
#endif

// ---------------------------------------------------------------------------
// Kernel A: QKV projection (1x1 conv == channel GEMM), f32 accumulate.
// ---------------------------------------------------------------------------
__global__ __launch_bounds__(256) void qkv_proj_kernel(
    const float* __restrict__ x, const float* __restrict__ y,
    const float* __restrict__ w_qkv,
    float* __restrict__ qf, float* __restrict__ kf, _Float16* __restrict__ vh)
{
  __shared__ float x_lds[CHAN * 128];
  const int mat = blockIdx.x / 64;        // 0=q, 1=k, 2=v
  const int rem = blockIdx.x % 64;
  const int b   = rem >> 5;
  const int nt  = (rem & 31) << 7;        // n-tile base
  const int tid = threadIdx.x;

  const float* in = ((mat == 0) ? x : y) + (size_t)b * CHAN * NSP;
  #pragma unroll
  for (int i = 0; i < 16; ++i) {
    const int idx = (i * 256 + tid) * 4;
    const int c = idx >> 7, n = idx & 127;
    *(float4*)&x_lds[idx] = *(const float4*)&in[(size_t)c * NSP + nt + n];
  }
  __syncthreads();

  const int n = tid & 127;
  const int ocbase = (tid >> 7) * 64;
  const float* wm = w_qkv + (size_t)mat * CHAN * CHAN;

  for (int og = 0; og < 64; og += 4) {
    const int oc = ocbase + og;
    float a0 = 0.f, a1 = 0.f, a2 = 0.f, a3 = 0.f;
    #pragma unroll 4
    for (int c = 0; c < CHAN; ++c) {
      const float xv = x_lds[c * 128 + n];
      a0 += wm[(oc + 0) * CHAN + c] * xv;
      a1 += wm[(oc + 1) * CHAN + c] * xv;
      a2 += wm[(oc + 2) * CHAN + c] * xv;
      a3 += wm[(oc + 3) * CHAN + c] * xv;
    }
    const size_t o = (size_t)(b * CHAN + oc) * NSP + nt + n;
    if (mat == 0) {
      qf[o] = a0; qf[o + NSP] = a1; qf[o + 2 * NSP] = a2; qf[o + 3 * NSP] = a3;
    } else if (mat == 1) {
      kf[o] = a0; kf[o + NSP] = a1; kf[o + 2 * NSP] = a2; kf[o + 3 * NSP] = a3;
    } else {
      vh[o]           = (_Float16)a0; vh[o + NSP]     = (_Float16)a1;
      vh[o + 2 * NSP] = (_Float16)a2; vh[o + 3 * NSP] = (_Float16)a3;
    }
  }
}

// ---------------------------------------------------------------------------
// Kernel B: per-row sumsq + l2-normalize along spatial axis, convert to f16.
// Q gets 10*log2(e) folded in; Q/K written token-major ([bh][n][32d]).
// ---------------------------------------------------------------------------
__global__ __launch_bounds__(256) void norm_kernel(
    const float* __restrict__ qf, const float* __restrict__ kf,
    _Float16* __restrict__ qh, _Float16* __restrict__ kh)
{
  __shared__ float red[256];
  const int mat = blockIdx.x >> 8;        // 0=q, 1=k
  const int row = blockIdx.x & 255;       // b*128 + oc
  const float* src = (mat == 0) ? qf : kf;
  _Float16* dst = (mat == 0) ? qh : kh;
  const size_t base = (size_t)row * NSP;
  const size_t tbase = (size_t)(row >> 5) * NSP * DHEAD + (row & 31);

  float v[16]; float ss = 0.f;
  #pragma unroll
  for (int i = 0; i < 16; ++i) {
    v[i] = src[base + i * 256 + threadIdx.x];
    ss += v[i] * v[i];
  }
  red[threadIdx.x] = ss;
  __syncthreads();
  for (int s = 128; s > 0; s >>= 1) {
    if (threadIdx.x < s) red[threadIdx.x] += red[threadIdx.x + s];
    __syncthreads();
  }
  const float scale = ((mat == 0) ? (10.f * LOG2E) : 1.f) /
                      fmaxf(sqrtf(red[0]), 1e-12f);
  #pragma unroll
  for (int i = 0; i < 16; ++i) {
    const int n = i * 256 + threadIdx.x;
    dst[tbase + (size_t)n * DHEAD] = (_Float16)(v[i] * scale);
  }
}

// ---------------------------------------------------------------------------
// Kernel C: flash attention, transposed formulation, double-buffered K/V
// staging. On toolchains exposing the TDM builtin, wave 0 drives the Tensor
// Data Mover (tensor_load_to_lds + s_wait_tensorcnt) to fill the next LDS
// buffer while all waves run WMMAs on the current one; otherwise a software-
// pipelined b128 copy does the same.
// ---------------------------------------------------------------------------
__global__ __launch_bounds__(256) void attn_kernel(
    const _Float16* __restrict__ qh, const _Float16* __restrict__ kh,
    const _Float16* __restrict__ vh, float* __restrict__ Obuf)
{
  __shared__ _Float16 k_lds[2][64 * DHEAD];   // [j][d] token-major, 2 x 4 KB
  __shared__ _Float16 v_lds[2][DHEAD * 64];   // [d][j],              2 x 4 KB

  const int bh   = blockIdx.x >> 5;
  const int qblk = (blockIdx.x & 31) << 7;
  const int wave = threadIdx.x >> 5;
  const int lane = threadIdx.x & 31;
  const int tid  = threadIdx.x;
  const int m    = lane & 15;
  const int hi   = lane >> 4;

  const _Float16* qp = qh + (size_t)bh * NSP * DHEAD;   // [n][d]
  const _Float16* kp = kh + (size_t)bh * NSP * DHEAD;   // [n][d]
  const _Float16* vp = vh + (size_t)bh * DHEAD * NSP;   // [d][n]
  float* op = Obuf + (size_t)bh * DHEAD * NSP;          // [d][n]

  const int qrow = qblk + wave * 16 + m;
  const uint4* qg4 = (const uint4*)qp;
  const v16h b_q = mk16(qg4[qrow * 4 + hi], qg4[qrow * 4 + 2 + hi]);

  float mrun = -1e30f, lrun = 0.f;
  v8f o0 = {}, o1 = {};

  // ---- prologue: fill buffer 0 with chunk 0 ----
#if USE_TDM
  if (wave == 0) {
    tdm_load_2d(kp, (unsigned)(uintptr_t)(void*)&k_lds[0][0], 64, DHEAD, DHEAD);
    tdm_load_2d(vp, (unsigned)(uintptr_t)(void*)&v_lds[0][0], DHEAD, 64, NSP);
    __builtin_amdgcn_s_wait_tensorcnt((short)0);
  }
#else
  {
    const int e = tid * 8;
    const int d = e >> 6, j = e & 63;
    *(uint4*)&k_lds[0][e] = *(const uint4*)&kp[e];
    *(uint4*)&v_lds[0][e] = *(const uint4*)&vp[(size_t)d * NSP + j];
  }
#endif
  __syncthreads();

  for (int it = 0; it < NSP / 64; ++it) {
    const int cur = it & 1;
    const int jn  = ((it + 1) * 64) & (NSP - 1);   // next chunk (wraps)

    // ---- kick off prefetch of the next chunk into the other buffer ----
#if USE_TDM
    if (wave == 0 && it + 1 < NSP / 64) {
      tdm_load_2d(kp + (size_t)jn * DHEAD,
                  (unsigned)(uintptr_t)(void*)&k_lds[cur ^ 1][0], 64, DHEAD, DHEAD);
      tdm_load_2d(vp + jn,
                  (unsigned)(uintptr_t)(void*)&v_lds[cur ^ 1][0], DHEAD, 64, NSP);
    }
#else
    uint4 kx, vx;
    {
      const int e = tid * 8;
      const int d = e >> 6, j = e & 63;
      kx = *(const uint4*)&kp[(size_t)jn * DHEAD + e];
      vx = *(const uint4*)&vp[(size_t)d * NSP + jn + j];
    }
#endif

    const uint4* kl4 = (const uint4*)&k_lds[cur][0];
    const uint4* vl4 = (const uint4*)&v_lds[cur][0];

    // ---- S^T tiles: A_k = 2 x ds_load_b128 ----
    v8f st[4];
    #pragma unroll
    for (int t = 0; t < 4; ++t) {
      const int j = t * 16 + m;
      const v16h a_k = mk16(kl4[j * 4 + hi], kl4[j * 4 + 2 + hi]);
      const v8f cz = {};
      st[t] = __builtin_amdgcn_wmma_f32_16x16x32_f16(false, a_k, false, b_q,
                                                     (short)0, cz, false, false);
    }

    // ---- online softmax over keys (exp2 domain) ----
    float tmax = st[0][0];
    #pragma unroll
    for (int t = 0; t < 4; ++t) {
      #pragma unroll
      for (int r = 0; r < 8; ++r) tmax = fmaxf(tmax, st[t][r]);
    }
    tmax = fmaxf(tmax, __shfl_xor(tmax, 16, 32));
    const float mnew  = fmaxf(mrun, tmax);
    const float alpha = exp2f(mrun - mnew);
    mrun = mnew;

    unsigned int pk[4][4];
    float rs = 0.f;
    #pragma unroll
    for (int t = 0; t < 4; ++t) {
      #pragma unroll
      for (int i = 0; i < 4; ++i) {
        const float p0 = exp2f(st[t][2 * i]     - mnew);
        const float p1 = exp2f(st[t][2 * i + 1] - mnew);
        rs += p0 + p1;
        pk[t][i] = pack2h(p0, p1);
      }
    }
    rs += __shfl_xor(rs, 16, 32);
    lrun = lrun * alpha + rs;
    #pragma unroll
    for (int r = 0; r < 8; ++r) { o0[r] *= alpha; o1[r] *= alpha; }

    // ---- build P^T B-operands: one packed exchange with lane^16 ----
    v8u bp0u, bp1u;
    #pragma unroll
    for (int i = 0; i < 4; ++i) {
      const unsigned int x0 = hi ? pk[0][i] : pk[1][i];
      const unsigned int y0 = (unsigned int)__shfl_xor((int)x0, 16, 32);
      const unsigned int x1 = hi ? pk[2][i] : pk[3][i];
      const unsigned int y1 = (unsigned int)__shfl_xor((int)x1, 16, 32);
      bp0u[i]     = hi ? y0       : pk[0][i];
      bp0u[i + 4] = hi ? pk[1][i] : y0;
      bp1u[i]     = hi ? y1       : pk[2][i];
      bp1u[i + 4] = hi ? pk[3][i] : y1;
    }
    const v16h bp0 = as_v16h(bp0u);
    const v16h bp1 = as_v16h(bp1u);

    // ---- O^T accumulation: A_v = 2 x b128 LDS loads per WMMA ----
    #pragma unroll
    for (int g = 0; g < 2; ++g) {
      v8f& og = g ? o1 : o0;
      #pragma unroll
      for (int c = 0; c < 2; ++c) {
        const int rbase = (16 * g + m) * 8 + 4 * c + 2 * hi;
        const v16h a_v = mk16(vl4[rbase], vl4[rbase + 1]);
        og = __builtin_amdgcn_wmma_f32_16x16x32_f16(false, a_v, false,
                                                    (c ? bp1 : bp0),
                                                    (short)0, og, false, false);
      }
    }

    // ---- commit prefetch, single barrier per iteration ----
#if USE_TDM
    if (wave == 0) __builtin_amdgcn_s_wait_tensorcnt((short)0);
#else
    {
      const int e = tid * 8;
      *(uint4*)&k_lds[cur ^ 1][e] = kx;
      *(uint4*)&v_lds[cur ^ 1][e] = vx;
    }
#endif
    __syncthreads();
  }

  // epilogue: O^T C-layout -> Obuf (b*h, d, n), divide by row sums
  const float inv = 1.f / lrun;
  const int nq = qblk + wave * 16 + m;
  #pragma unroll
  for (int r = 0; r < 8; ++r) {
    const int d = r + 8 * hi;
    op[(size_t)d * NSP + nq]        = o0[r] * inv;
    op[(size_t)(16 + d) * NSP + nq] = o1[r] * inv;
  }
}

// ---------------------------------------------------------------------------
// Kernel D: output projection (1x1 conv) + bias -> d_out (f32), LDS-tiled.
// ---------------------------------------------------------------------------
__global__ __launch_bounds__(256) void out_proj_kernel(
    const float* __restrict__ Obuf, const float* __restrict__ w_out,
    const float* __restrict__ b_out, float* __restrict__ out)
{
  __shared__ float o_lds[HID * 128];
  const int b   = blockIdx.x >> 5;
  const int nt  = (blockIdx.x & 31) << 7;
  const int tid = threadIdx.x;

  const float* in = Obuf + (size_t)b * HID * NSP;
  #pragma unroll
  for (int i = 0; i < 16; ++i) {
    const int idx = (i * 256 + tid) * 4;
    const int c = idx >> 7, n = idx & 127;
    *(float4*)&o_lds[idx] = *(const float4*)&in[(size_t)c * NSP + nt + n];
  }
  __syncthreads();

  const int n = tid & 127;
  const int obase = (tid >> 7) * 64;
  for (int og = 0; og < 64; og += 4) {
    const int o = obase + og;
    float a0 = b_out[o], a1 = b_out[o + 1], a2 = b_out[o + 2], a3 = b_out[o + 3];
    #pragma unroll 4
    for (int c = 0; c < HID; ++c) {
      const float xv = o_lds[c * 128 + n];
      a0 += w_out[(o + 0) * HID + c] * xv;
      a1 += w_out[(o + 1) * HID + c] * xv;
      a2 += w_out[(o + 2) * HID + c] * xv;
      a3 += w_out[(o + 3) * HID + c] * xv;
    }
    const size_t oo = (size_t)(b * CHAN + o) * NSP + nt + n;
    out[oo] = a0; out[oo + NSP] = a1; out[oo + 2 * NSP] = a2; out[oo + 3 * NSP] = a3;
  }
}

// ---------------------------------------------------------------------------
extern "C" void kernel_launch(void* const* d_in, const int* in_sizes, int n_in,
                              void* d_out, int out_size, void* d_ws, size_t ws_size,
                              hipStream_t stream) {
  (void)in_sizes; (void)n_in; (void)out_size; (void)ws_size;
  const float* x     = (const float*)d_in[0];
  const float* y     = (const float*)d_in[1];
  const float* w_qkv = (const float*)d_in[2];
  const float* w_out = (const float*)d_in[3];
  const float* b_out = (const float*)d_in[4];
  float* out = (float*)d_out;

  char* ws = (char*)d_ws;
  float*    qf   = (float*)(ws);                                  // 4 MB
  float*    kf   = (float*)(ws + (size_t)(4u  << 20));            // 4 MB
  _Float16* qh   = (_Float16*)(ws + (size_t)(8u  << 20));         // 2 MB [n][d]
  _Float16* kh   = (_Float16*)(ws + (size_t)(10u << 20));         // 2 MB [n][d]
  _Float16* vh   = (_Float16*)(ws + (size_t)(12u << 20));         // 2 MB [d][n]
  float*    Obuf = (float*)(ws + (size_t)(14u << 20));            // 4 MB

  qkv_proj_kernel<<<192, 256, 0, stream>>>(x, y, w_qkv, qf, kf, vh);
  norm_kernel<<<512, 256, 0, stream>>>(qf, kf, qh, kh);
  attn_kernel<<<256, 256, 0, stream>>>(qh, kh, vh, Obuf);
  out_proj_kernel<<<64, 256, 0, stream>>>(Obuf, w_out, b_out, out);
}
